// GNN_27934467293571
// MI455X (gfx1250) — matrix operand order repaired and checked by reading, not compiled, not run on hardware
//
#include <hip/hip_runtime.h>
#include <math.h>

typedef __attribute__((ext_vector_type(16))) _Float16 v16h;
typedef __attribute__((ext_vector_type(8)))  float    v8f;

#define DIM 128
#define NEG_SLOPE 0.2f
#define BN_EPS 1e-5f

// ---------------------------------------------------------------------------
// float atomic max via sign-split int/uint ordering trick (init = -inf)
// ---------------------------------------------------------------------------
__device__ __forceinline__ void atomicMaxF(float* addr, float v) {
    if (v >= 0.0f) atomicMax((int*)addr, __float_as_int(v));
    else           atomicMin((unsigned int*)addr, (unsigned int)__float_as_int(v));
}

// ---------------------------------------------------------------------------
// fill
// ---------------------------------------------------------------------------
__global__ void k_fill(float* __restrict__ p, int n, float v) {
    int i = blockIdx.x * blockDim.x + threadIdx.x;
    if (i < n) p[i] = v;
}

// ---------------------------------------------------------------------------
// Weight prep: W [128x128] row-major (k,n) fp32 -> transposed (n,k) f16 hi/lo
// ---------------------------------------------------------------------------
__global__ void k_prepW(const float* __restrict__ W,
                        _Float16* __restrict__ Wth, _Float16* __restrict__ Wtl) {
    int idx = blockIdx.x * 256 + threadIdx.x;     // 16384 threads
    int k = idx >> 7, n = idx & 127;
    float w = W[idx];
    _Float16 h = (_Float16)w;
    Wth[n * DIM + k] = h;
    Wtl[n * DIM + k] = (_Float16)(w - (float)h);
}

// ---------------------------------------------------------------------------
// GEMM: XS[N,128] = X[N,128] @ W[128,128] using f16 hi/lo split WMMA.
// Block = 256 threads = 8 waves; block handles 16 rows, wave w -> cols [16w,16w+16).
// A fragment (16x32 f16): lane(l=lane%16, g=lane/16) holds row M=l,
//   element e -> K = k0 + g*16 + e
// B fragment (32x16 f16): lane(l,g) holds col N=l, element e -> K = k0 + g*16 + e
//   (contiguous in the pre-transposed (n,k) weight array).
// ---------------------------------------------------------------------------
__global__ void __launch_bounds__(256)
k_gemm_wmma(const float* __restrict__ X,
            const _Float16* __restrict__ Bh, const _Float16* __restrict__ Bl,
            float* __restrict__ XS, int nrows) {
    int wave = threadIdx.x >> 5;
    int lane = threadIdx.x & 31;
    int l = lane & 15, g = lane >> 4;
    int row0 = blockIdx.x << 4;
    int col0 = wave << 4;

    int arow = row0 + l; if (arow >= nrows) arow = nrows - 1;
    const float*    rp  = X  + (size_t)arow * DIM;
    const _Float16* bph = Bh + (size_t)(col0 + l) * DIM;
    const _Float16* bpl = Bl + (size_t)(col0 + l) * DIM;

    v8f acc = {};
#pragma unroll
    for (int kt = 0; kt < 4; ++kt) {
        int k0 = kt * 32;
        float f[16];
        const float4* c1 = (const float4*)(rp + k0 + g * 16);
        const float4* c2 = (const float4*)(rp + k0 + g * 16 + 8);
        float4 t;
        t = c1[0]; f[0]  = t.x; f[1]  = t.y; f[2]  = t.z; f[3]  = t.w;
        t = c1[1]; f[4]  = t.x; f[5]  = t.y; f[6]  = t.z; f[7]  = t.w;
        t = c2[0]; f[8]  = t.x; f[9]  = t.y; f[10] = t.z; f[11] = t.w;
        t = c2[1]; f[12] = t.x; f[13] = t.y; f[14] = t.z; f[15] = t.w;

        v16h ah, al;
#pragma unroll
        for (int j = 0; j < 16; ++j) {
            _Float16 h = (_Float16)f[j];
            ah[j] = h;
            al[j] = (_Float16)(f[j] - (float)h);
        }
        v16h bh = *(const v16h*)(bph + k0 + g * 16);
        v16h bl = *(const v16h*)(bpl + k0 + g * 16);

        acc = __builtin_amdgcn_wmma_f32_16x16x32_f16(false, ah, false, bh, (short)0, acc, false, false);
        acc = __builtin_amdgcn_wmma_f32_16x16x32_f16(false, ah, false, bl, (short)0, acc, false, false);
        acc = __builtin_amdgcn_wmma_f32_16x16x32_f16(false, al, false, bh, (short)0, acc, false, false);
    }

    // C/D layout: lane(l,g), VGPR r -> D[m = r + 8g][n = l]
    float* op = XS + (size_t)row0 * DIM + col0 + l;
#pragma unroll
    for (int r = 0; r < 8; ++r) {
        int m = r + 8 * g;
        if (row0 + m < nrows) op[(size_t)m * DIM] = acc[r];
    }
}

// ---------------------------------------------------------------------------
// Per-node attention dots: asrc[i] = xs[i].a_src ; adst[i] = xs[i].a_dst
// Wave per node; lane handles 4 channels; shfl reduction.
// ---------------------------------------------------------------------------
__global__ void k_alpha(const float* __restrict__ XS,
                        const float* __restrict__ a_src, const float* __restrict__ a_dst,
                        float* __restrict__ outs, float* __restrict__ outd, int n) {
    int wave = threadIdx.x >> 5, lane = threadIdx.x & 31;
    int node = blockIdx.x * 8 + wave;
    if (node >= n) return;
    float4 xv = ((const float4*)(XS + (size_t)node * DIM))[lane];
    float4 as = ((const float4*)a_src)[lane];
    float4 ad = ((const float4*)a_dst)[lane];
    float s = xv.x * as.x + xv.y * as.y + xv.z * as.z + xv.w * as.w;
    float d = xv.x * ad.x + xv.y * ad.y + xv.z * ad.z + xv.w * ad.w;
#pragma unroll
    for (int off = 16; off > 0; off >>= 1) {
        s += __shfl_down(s, off);
        d += __shfl_down(d, off);
    }
    if (lane == 0) { outs[node] = s; outd[node] = d; }
}

// ---------------------------------------------------------------------------
// Edge logits + segment max  (edges E..E+n-1 are implicit self-loops)
// ---------------------------------------------------------------------------
__global__ void k_logits(const int* __restrict__ src, const int* __restrict__ dst,
                         const float* __restrict__ as, const float* __restrict__ ad,
                         float* __restrict__ logits, float* __restrict__ mbuf, int E, int n) {
    int e = blockIdx.x * 256 + threadIdx.x;
    if (e >= E + n) return;
    int s, d;
    if (e < E) { s = src[e]; d = dst[e]; } else { s = d = e - E; }
    float v = as[s] + ad[d];
    v = v > 0.0f ? v : NEG_SLOPE * v;
    logits[e] = v;
    atomicMaxF(mbuf + d, v);
}

// ---------------------------------------------------------------------------
// exp(logit - max[dst]) + segment sum (exp value overwrites logits)
// ---------------------------------------------------------------------------
__global__ void k_expsum(const int* __restrict__ dst, float* __restrict__ logits,
                         const float* __restrict__ mbuf, float* __restrict__ denom, int E, int n) {
    int e = blockIdx.x * 256 + threadIdx.x;
    if (e >= E + n) return;
    int d = (e < E) ? dst[e] : e - E;
    float v = expf(logits[e] - mbuf[d]);
    logits[e] = v;
    atomicAdd(denom + d, v);
}

// ---------------------------------------------------------------------------
// Aggregation: OUT[dst] += (w[e]/denom[dst]) * XS[src]   (wave per edge)
// ---------------------------------------------------------------------------
__global__ void k_agg(const int* __restrict__ src, const int* __restrict__ dst,
                      const float* __restrict__ w, const float* __restrict__ denom,
                      const float* __restrict__ XS, float* __restrict__ OUT, int E, int n) {
    int wave = threadIdx.x >> 5, lane = threadIdx.x & 31;
    int e = blockIdx.x * 8 + wave;
    if (e >= E + n) return;
    int s, d;
    if (e < E) { s = src[e]; d = dst[e]; } else { s = d = e - E; }
    float alpha = w[e] / (denom[d] + 1e-16f);
    float4 xv = ((const float4*)(XS + (size_t)s * DIM))[lane];
    float* op = OUT + (size_t)d * DIM + lane * 4;
    atomicAdd(op + 0, alpha * xv.x);
    atomicAdd(op + 1, alpha * xv.y);
    atomicAdd(op + 2, alpha * xv.z);
    atomicAdd(op + 3, alpha * xv.w);
}

// ---------------------------------------------------------------------------
// BN stats: per-channel sum / sumsq of (H + bias). 128 threads = 128 channels.
// ---------------------------------------------------------------------------
__global__ void k_bnstats(const float* __restrict__ H, const float* __restrict__ bias,
                          float* __restrict__ sum, float* __restrict__ sumsq, int n) {
    int c = threadIdx.x;
    float b = bias[c];
    float s = 0.0f, s2 = 0.0f;
    for (int r = blockIdx.x; r < n; r += gridDim.x) {
        float v = H[(size_t)r * DIM + c] + b;
        s += v; s2 += v * v;
    }
    atomicAdd(sum + c, s);
    atomicAdd(sumsq + c, s2);
}

__global__ void k_bnfinal(const float* __restrict__ sum, const float* __restrict__ sumsq,
                          const float* __restrict__ gamma, const float* __restrict__ beta,
                          float* __restrict__ sc, float* __restrict__ sh, int n) {
    int c = threadIdx.x;
    float inv = 1.0f / (float)n;
    float mu = sum[c] * inv;
    float var = sumsq[c] * inv - mu * mu;
    float scale = gamma[c] * rsqrtf(var + BN_EPS);
    sc[c] = scale;
    sh[c] = beta[c] - mu * scale;
}

__global__ void k_bnapply(const float* __restrict__ in, const float* __restrict__ bias,
                          const float* __restrict__ sc, const float* __restrict__ sh,
                          float* __restrict__ out, int total, int doRelu) {
    int i = blockIdx.x * 256 + threadIdx.x;
    if (i >= total) return;
    int c = i & 127;
    float v = (in[i] + bias[c]) * sc[c] + sh[c];
    if (doRelu) v = fmaxf(v, 0.0f);
    out[i] = v;
}

// ---------------------------------------------------------------------------
// Orchestration
// ---------------------------------------------------------------------------
extern "C" void kernel_launch(void* const* d_in, const int* in_sizes, int n_in,
                              void* d_out, int out_size, void* d_ws, size_t ws_size,
                              hipStream_t stream) {
    const float* x  = (const float*)d_in[0];
    const int*   ei = (const int*)d_in[1];
    const float* Wm[2]   = { (const float*)d_in[2],  (const float*)d_in[8]  };
    const float* asV[2]  = { (const float*)d_in[3],  (const float*)d_in[9]  };
    const float* adV[2]  = { (const float*)d_in[4],  (const float*)d_in[10] };
    const float* bV[2]   = { (const float*)d_in[5],  (const float*)d_in[11] };
    const float* gV[2]   = { (const float*)d_in[6],  (const float*)d_in[12] };
    const float* beV[2]  = { (const float*)d_in[7],  (const float*)d_in[13] };

    const int N  = in_sizes[0] / DIM;
    const int E  = in_sizes[1] / 2;
    const int Et = E + N;
    const int* src = ei;
    const int* dst = ei + E;

    char* p = (char*)d_ws;
    auto carve = [&](size_t bytes) -> void* {
        void* r = (void*)p;
        p += (bytes + 255) & ~(size_t)255;
        return r;
    };
    float*    xs     = (float*)carve((size_t)N * DIM * 4);
    float*    agg    = (float*)carve((size_t)N * DIM * 4);
    float*    logits = (float*)carve((size_t)Et * 4);
    float*    asrc   = (float*)carve((size_t)N * 4);
    float*    adst   = (float*)carve((size_t)N * 4);
    float*    mbuf   = (float*)carve((size_t)N * 4);
    float*    denom  = (float*)carve((size_t)N * 4);
    float*    sums   = (float*)carve(DIM * 4);
    float*    sumsq  = (float*)carve(DIM * 4);
    float*    sc     = (float*)carve(DIM * 4);
    float*    sh     = (float*)carve(DIM * 4);
    _Float16* Wth    = (_Float16*)carve(DIM * DIM * 2);
    _Float16* Wtl    = (_Float16*)carve(DIM * DIM * 2);

    float* out = (float*)d_out;
    const float* gemmIn[2] = { x, agg };
    float*       aggOut[2] = { agg, out };

    const int total   = N * DIM;
    const int nbTot   = (total + 255) / 256;
    const int nbNode  = (N + 255) / 256;
    const int nbEdge  = (Et + 255) / 256;
    const int nbWaveN = (N + 7) / 8;
    const int nbWaveE = (Et + 7) / 8;
    const int nbRows  = (N + 15) / 16;

    for (int L = 0; L < 2; ++L) {
        k_prepW   <<<64, 256, 0, stream>>>(Wm[L], Wth, Wtl);
        k_gemm_wmma<<<nbRows, 256, 0, stream>>>(gemmIn[L], Wth, Wtl, xs, N);
        k_alpha   <<<nbWaveN, 256, 0, stream>>>(xs, asV[L], adV[L], asrc, adst, N);

        k_fill<<<nbNode, 256, 0, stream>>>(mbuf, N, -INFINITY);
        k_fill<<<nbNode, 256, 0, stream>>>(denom, N, 0.0f);
        k_fill<<<nbTot,  256, 0, stream>>>(aggOut[L], total, 0.0f);
        k_fill<<<1, 128, 0, stream>>>(sums, DIM, 0.0f);
        k_fill<<<1, 128, 0, stream>>>(sumsq, DIM, 0.0f);

        k_logits<<<nbEdge, 256, 0, stream>>>(src, dst, asrc, adst, logits, mbuf, E, N);
        k_expsum<<<nbEdge, 256, 0, stream>>>(dst, logits, mbuf, denom, E, N);
        k_agg   <<<nbWaveE, 256, 0, stream>>>(src, dst, logits, denom, xs, aggOut[L], E, N);

        k_bnstats<<<512, 128, 0, stream>>>(aggOut[L], bV[L], sums, sumsq, N);
        k_bnfinal<<<1, 128, 0, stream>>>(sums, sumsq, gV[L], beV[L], sc, sh, N);
        k_bnapply<<<nbTot, 256, 0, stream>>>(aggOut[L], bV[L], sc, sh, aggOut[L], total, (L == 0) ? 1 : 0);
    }
}
